// CasualSelfAttention_85907935855356
// MI455X (gfx1250) — matrix operand order repaired and checked by reading, not compiled, not run on hardware
//
#include <hip/hip_runtime.h>

typedef __bf16 bf16_t;
typedef __attribute__((ext_vector_type(16))) __bf16 v16bf;
typedef __attribute__((ext_vector_type(8)))  __bf16 v8bf;
typedef __attribute__((ext_vector_type(8)))  float  v8f;

#define B_  2
#define S_  2048
#define D_  1024
#define H_  16
#define DK  64
#define NEG_INF (-1.0e30f)

// ---------- WMMA helpers ----------

__device__ __forceinline__ v8f wmma_bf16(v16bf a, v16bf b, v8f c) {
  // v_wmma_f32_16x16x32_bf16  (D = A*B + C), f32 accumulate
  return __builtin_amdgcn_wmma_f32_16x16x32_bf16(
      /*neg_a=*/false, a, /*neg_b=*/false, b,
      /*c_mod=*/(short)0, c, /*reuse_a=*/false, /*reuse_b=*/false);
}

// A-matrix fragment (16 rows x 32 K), row-major source, ld in elements.
// Lane l<16 : row=l,    K = {0..7, 16..23}
// Lane l>=16: row=l-16, K = {8..15, 24..31}
__device__ __forceinline__ v16bf load_a_frag(const bf16_t* __restrict__ p0, int ld) {
  const int lane = threadIdx.x & 31;
  const bf16_t* p = p0 + (size_t)(lane & 15) * ld + ((lane >> 4) << 3);
  v8bf lo = *reinterpret_cast<const v8bf*>(p);        // K offsets +0..7
  v8bf hi = *reinterpret_cast<const v8bf*>(p + 16);   // K offsets +16..23
  v16bf r;
#pragma unroll
  for (int i = 0; i < 8; ++i) { r[i] = lo[i]; r[i + 8] = hi[i]; }
  return r;
}

// B-matrix fragment (32 K x 16 cols). Source row n holds column n of B,
// contiguous along K.  Lanes 0-15: col=lane, K={0..7,16..23}; lanes 16-31: K={8..15,24..31}.
__device__ __forceinline__ v16bf load_b_frag(const bf16_t* __restrict__ p0, int ld) {
  const int lane = threadIdx.x & 31;
  const bf16_t* p = p0 + (size_t)(lane & 15) * ld + ((lane >> 4) << 3);
  v8bf lo = *reinterpret_cast<const v8bf*>(p);
  v8bf hi = *reinterpret_cast<const v8bf*>(p + 16);
  v16bf r;
#pragma unroll
  for (int i = 0; i < 8; ++i) { r[i] = lo[i]; r[i + 8] = hi[i]; }
  return r;
}

__device__ __forceinline__ float rowmax16(float v) {
  v = fmaxf(v, __shfl_xor(v, 1));
  v = fmaxf(v, __shfl_xor(v, 2));
  v = fmaxf(v, __shfl_xor(v, 4));
  v = fmaxf(v, __shfl_xor(v, 8));
  return v;
}
__device__ __forceinline__ float rowsum16(float v) {
  v += __shfl_xor(v, 1);
  v += __shfl_xor(v, 2);
  v += __shfl_xor(v, 4);
  v += __shfl_xor(v, 8);
  return v;
}

// ---------- Kernel 1: fp32 -> bf16 convert ----------

__global__ void cvt_f32_bf16(const float* __restrict__ src, bf16_t* __restrict__ dst, int n) {
  int i = blockIdx.x * blockDim.x + threadIdx.x;
  if (i < n) dst[i] = (bf16_t)src[i];
}

// ---------- Kernel 2: fused QKV projection + RoPE (64x32 tile per wave) ----
// 4x2 register blocking: acc = 64 VGPRs, frags = 48 VGPRs -> no spills at the
// 2-waves/SIMD VGPR budget.  Per K-step: 6 KB fragments, 8 WMMAs -> ~22 FLOP/B.
// q,k get RoPE, stored [b,h,s,d]; v stored transposed [b,h,d,s].

__global__ __launch_bounds__(256, 2)
void qkv_rope_kernel(const bf16_t* __restrict__ xb,
                     const bf16_t* __restrict__ wqb,
                     const bf16_t* __restrict__ wkb,
                     const bf16_t* __restrict__ wvb,
                     const int*    __restrict__ pos,
                     bf16_t* __restrict__ Q,
                     bf16_t* __restrict__ K,
                     bf16_t* __restrict__ Vt) {
  const int wave = (blockIdx.x * blockDim.x + threadIdx.x) >> 5;
  const int lane = threadIdx.x & 31;
  const int ntM = (B_ * S_) / 64;   // 64
  const int ntN = D_ / 32;          // 32
  const int proj = wave / (ntM * ntN);
  const int rem  = wave % (ntM * ntN);
  const int row0 = (rem / ntN) * 64;
  const int col0 = (rem % ntN) * 32;
  const bf16_t* w = (proj == 0) ? wqb : (proj == 1) ? wkb : wvb;

  v8f acc[4][2] = {};
  for (int k = 0; k < D_; k += 32) {
    v16bf a[4], b[2];
#pragma unroll
    for (int mi = 0; mi < 4; ++mi)
      a[mi] = load_a_frag(xb + (size_t)(row0 + mi * 16) * D_ + k, D_);
#pragma unroll
    for (int ni = 0; ni < 2; ++ni)
      b[ni] = load_b_frag(w + (size_t)(col0 + ni * 16) * D_ + k, D_);
#pragma unroll
    for (int mi = 0; mi < 4; ++mi)
#pragma unroll
      for (int ni = 0; ni < 2; ++ni)
        acc[mi][ni] = wmma_bf16(a[mi], b[ni], acc[mi][ni]);
  }

  const int half = lane >> 4;
  const int n    = lane & 15;

#pragma unroll
  for (int mi = 0; mi < 4; ++mi) {
#pragma unroll
    for (int ni = 0; ni < 2; ++ni) {
      const int o = col0 + ni * 16 + n;   // output feature
      const int h = o >> 6;
      const int d = o & 63;
#pragma unroll
      for (int r = 0; r < 8; ++r) {
        const int m  = row0 + mi * 16 + r + half * 8;   // flattened (b,s)
        const int bb = m / S_;
        const int s  = m % S_;
        float v = acc[mi][ni][r];
        if (proj == 2) {
          Vt[(((size_t)(bb * H_ + h)) * DK + d) * S_ + s] = (bf16_t)v;
        } else {
          const float p  = (float)pos[bb * S_ + s];
          const int   de = d & ~1;
          const float freq = __expf(-(float)de * 0.14391157f); // theta^{-de/64}
          float sn, cs;
          __sincosf(p * freq, &sn, &cs);
          const float partner = __shfl_xor(v, 1);
          float res;
          if ((d & 1) == 0) res = v * cs - partner * sn;   // xe*cos - xo*sin
          else              res = partner * sn + v * cs;   // xe*sin + xo*cos
          bf16_t* dst = (proj == 0) ? Q : K;
          dst[(((size_t)(bb * H_ + h)) * S_ + s) * DK + d] = (bf16_t)res;
        }
      }
    }
  }
}

// ---------- Kernel 3: causal flash attention (32 queries per wave) --------
// Two stacked 16-row q-tiles share every K- and V-fragment load (2x reuse).
// Online softmax in WMMA C-layout; P tiles relayout C->A through per-wave
// LDS with split-counter waits (s_wait_dscnt).

__global__ __launch_bounds__(256, 2)
void attn_kernel(const bf16_t* __restrict__ Q,
                 const bf16_t* __restrict__ K,
                 const bf16_t* __restrict__ Vt,
                 bf16_t* __restrict__ O) {
  __shared__ __align__(32) bf16_t pbuf[8][2 * 16 * 32];   // 2 KB per wave

  const int gwave = (blockIdx.x * blockDim.x + threadIdx.x) >> 5;
  const int lwave = threadIdx.x >> 5;
  const int lane  = threadIdx.x & 31;
  const int half  = lane >> 4;
  const int n     = lane & 15;

  const int nqb = S_ / 32;               // 64 query blocks of 32
  const int qb  = gwave % nqb;
  const int bh  = gwave / nqb;           // 0..31
  const int q0  = qb * 32;

  const bf16_t* Qp = Q  + (size_t)bh * S_ * DK;
  const bf16_t* Kp = K  + (size_t)bh * S_ * DK;
  const bf16_t* Vp = Vt + (size_t)bh * DK * S_;
  bf16_t* pb = &pbuf[lwave][0];

  // Q tiles as A fragments: qa[qtile][k-half]
  v16bf qa[2][2];
#pragma unroll
  for (int qt = 0; qt < 2; ++qt) {
    qa[qt][0] = load_a_frag(Qp + (size_t)(q0 + qt * 16) * DK + 0,  DK);
    qa[qt][1] = load_a_frag(Qp + (size_t)(q0 + qt * 16) * DK + 32, DK);
  }

  v8f acc[2][4] = {};
  float mrow[2][8], lrow[2][8];
#pragma unroll
  for (int qt = 0; qt < 2; ++qt)
#pragma unroll
    for (int r = 0; r < 8; ++r) { mrow[qt][r] = NEG_INF; lrow[qt][r] = 0.0f; }

  const int kb_end = q0 / 32 + 1;        // causal: key blocks of 32
  const float scale = 0.125f;            // 1/sqrt(64)

  for (int kb = 0; kb < kb_end; ++kb) {
    // ---- scores: K-fragments loaded once, shared by both q tiles ----
    v16bf kf[2][2];                      // kf[ktile][k-half]
#pragma unroll
    for (int kt = 0; kt < 2; ++kt) {
      kf[kt][0] = load_b_frag(Kp + (size_t)(kb * 32 + kt * 16) * DK + 0,  DK);
      kf[kt][1] = load_b_frag(Kp + (size_t)(kb * 32 + kt * 16) * DK + 32, DK);
    }
    v8f s[2][2] = {};                    // s[qtile][ktile]
#pragma unroll
    for (int qt = 0; qt < 2; ++qt)
#pragma unroll
      for (int kt = 0; kt < 2; ++kt) {
        s[qt][kt] = wmma_bf16(qa[qt][0], kf[kt][0], s[qt][kt]);
        s[qt][kt] = wmma_bf16(qa[qt][1], kf[kt][1], s[qt][kt]);
      }

    // ---- online softmax update (C-layout), per q tile ----
#pragma unroll
    for (int qt = 0; qt < 2; ++qt) {
#pragma unroll
      for (int r = 0; r < 8; ++r) {
        const int qi  = q0 + qt * 16 + r + half * 8;
        const int kj0 = kb * 32 + n;
        const int kj1 = kb * 32 + 16 + n;
        float v0 = (kj0 <= qi) ? s[qt][0][r] * scale : NEG_INF;
        float v1 = (kj1 <= qi) ? s[qt][1][r] * scale : NEG_INF;

        float mx   = rowmax16(fmaxf(v0, v1));
        float mnew = fmaxf(mrow[qt][r], mx);
        float corr = __expf(mrow[qt][r] - mnew);
        float p0   = __expf(v0 - mnew);
        float p1   = __expf(v1 - mnew);
        float ls   = rowsum16(p0 + p1);

        lrow[qt][r] = lrow[qt][r] * corr + ls;
        mrow[qt][r] = mnew;
#pragma unroll
        for (int dt = 0; dt < 4; ++dt) acc[qt][dt][r] *= corr;

        const int row = r + half * 8;
        pb[qt * 512 + row * 32 + n]      = (bf16_t)p0;
        pb[qt * 512 + row * 32 + 16 + n] = (bf16_t)p1;
      }
    }

    // per-wave LDS ordering: wait for DS stores before reading back
    asm volatile("s_wait_dscnt 0x0" ::: "memory");

    // ---- PV: V-fragments loaded once, shared by both q tiles ----
    v16bf pa[2];
    pa[0] = load_a_frag(pb,       32);
    pa[1] = load_a_frag(pb + 512, 32);
#pragma unroll
    for (int dt = 0; dt < 4; ++dt) {
      v16bf vb = load_b_frag(Vp + (size_t)(dt * 16) * S_ + kb * 32, S_);
      acc[0][dt] = wmma_bf16(pa[0], vb, acc[0][dt]);
      acc[1][dt] = wmma_bf16(pa[1], vb, acc[1][dt]);
    }
    asm volatile("s_wait_dscnt 0x0" ::: "memory");
  }

  // ---- normalize and store to [b, s, h*64 + d] (bf16) ----
  const int bb = bh >> 4;
  const int h  = bh & 15;
#pragma unroll
  for (int qt = 0; qt < 2; ++qt) {
#pragma unroll
    for (int r = 0; r < 8; ++r) {
      const float inv = 1.0f / lrow[qt][r];
      const int   s   = q0 + qt * 16 + r + half * 8;
      const size_t base = ((size_t)(bb * S_ + s)) * D_ + h * DK;
      O[base +  0 + n] = (bf16_t)(acc[qt][0][r] * inv);
      O[base + 16 + n] = (bf16_t)(acc[qt][1][r] * inv);
      O[base + 32 + n] = (bf16_t)(acc[qt][2][r] * inv);
      O[base + 48 + n] = (bf16_t)(acc[qt][3][r] * inv);
    }
  }
}

// ---------- Kernel 4: output projection (64x32 tile per wave, fp32 out) ----

__global__ __launch_bounds__(256, 2)
void out_proj_kernel(const bf16_t* __restrict__ A,
                     const bf16_t* __restrict__ Wo,
                     float* __restrict__ out) {
  const int wave = (blockIdx.x * blockDim.x + threadIdx.x) >> 5;
  const int lane = threadIdx.x & 31;
  const int ntN  = D_ / 32;              // 32
  const int row0 = (wave / ntN) * 64;
  const int col0 = (wave % ntN) * 32;

  v8f acc[4][2] = {};
  for (int k = 0; k < D_; k += 32) {
    v16bf a[4], b[2];
#pragma unroll
    for (int mi = 0; mi < 4; ++mi)
      a[mi] = load_a_frag(A + (size_t)(row0 + mi * 16) * D_ + k, D_);
#pragma unroll
    for (int ni = 0; ni < 2; ++ni)
      b[ni] = load_b_frag(Wo + (size_t)(col0 + ni * 16) * D_ + k, D_);
#pragma unroll
    for (int mi = 0; mi < 4; ++mi)
#pragma unroll
      for (int ni = 0; ni < 2; ++ni)
        acc[mi][ni] = wmma_bf16(a[mi], b[ni], acc[mi][ni]);
  }

  const int half = lane >> 4;
  const int n    = lane & 15;
#pragma unroll
  for (int mi = 0; mi < 4; ++mi)
#pragma unroll
    for (int ni = 0; ni < 2; ++ni)
#pragma unroll
      for (int r = 0; r < 8; ++r)
        out[(size_t)(row0 + mi * 16 + r + half * 8) * D_ + col0 + ni * 16 + n] =
            acc[mi][ni][r];
}

// ---------- launch ----------

extern "C" void kernel_launch(void* const* d_in, const int* in_sizes, int n_in,
                              void* d_out, int out_size, void* d_ws, size_t ws_size,
                              hipStream_t stream) {
  const float* x   = (const float*)d_in[0];
  const int*   pos = (const int*)  d_in[1];
  const float* wq  = (const float*)d_in[2];
  const float* wk  = (const float*)d_in[3];
  const float* wv  = (const float*)d_in[4];
  const float* wo  = (const float*)d_in[5];
  float* out = (float*)d_out;

  // workspace carve (bf16 buffers)
  size_t off = 0;
  auto carve = [&](size_t elems) {
    void* p = (char*)d_ws + off;
    off += (elems * sizeof(bf16_t) + 255) & ~(size_t)255;
    return (bf16_t*)p;
  };
  const size_t NX = (size_t)B_ * S_ * D_;       // 4M elems
  const size_t NW = (size_t)D_ * D_;            // 1M elems
  const size_t NH = (size_t)B_ * H_ * S_ * DK;  // 4M elems
  bf16_t* xb  = carve(NX);
  bf16_t* wqb = carve(NW);
  bf16_t* wkb = carve(NW);
  bf16_t* wvb = carve(NW);
  bf16_t* wob = carve(NW);
  bf16_t* Qb  = carve(NH);
  bf16_t* Kb  = carve(NH);
  bf16_t* Vtb = carve(NH);
  bf16_t* Ab  = carve(NX);

  // 1) convert to bf16
  cvt_f32_bf16<<<(int)((NX + 255) / 256), 256, 0, stream>>>(x,  xb,  (int)NX);
  cvt_f32_bf16<<<(int)((NW + 255) / 256), 256, 0, stream>>>(wq, wqb, (int)NW);
  cvt_f32_bf16<<<(int)((NW + 255) / 256), 256, 0, stream>>>(wk, wkb, (int)NW);
  cvt_f32_bf16<<<(int)((NW + 255) / 256), 256, 0, stream>>>(wv, wvb, (int)NW);
  cvt_f32_bf16<<<(int)((NW + 255) / 256), 256, 0, stream>>>(wo, wob, (int)NW);

  // 2) QKV projection + RoPE: 3 * 64 * 32 waves (64x32 tile each)
  qkv_rope_kernel<<<(3 * 64 * 32) / 8, 256, 0, stream>>>(xb, wqb, wkb, wvb, pos, Qb, Kb, Vtb);

  // 3) flash attention: 32 bh * 64 q-blocks = 2048 waves
  attn_kernel<<<2048 / 8, 256, 0, stream>>>(Qb, Kb, Vtb, Ab);

  // 4) output projection: 64 * 32 waves (64x32 tile each)
  out_proj_kernel<<<(64 * 32) / 8, 256, 0, stream>>>(Ab, wob, out);
}
